// TDTNet_9216999817556
// MI455X (gfx1250) — matrix operand order repaired and checked by reading, not compiled, not run on hardware
//
#include <hip/hip_runtime.h>

#define N_ATOMS 20000
#define N_EDGES 640000
#define DIM     128
#define NG      32
#define NH      8
#define NINTER  4
#define NT_NODE (N_ATOMS/16)   // 1250
#define NT_EDGE (N_EDGES/16)   // 40000
#define PI_F    3.14159265358979f

typedef __attribute__((ext_vector_type(16))) _Float16 v16h;
typedef __attribute__((ext_vector_type(8)))  _Float16 v8h;
typedef __attribute__((ext_vector_type(4)))  _Float16 v4h;
typedef __attribute__((ext_vector_type(8)))  float    v8f;
typedef __attribute__((ext_vector_type(4)))  float    v4f;
typedef __attribute__((ext_vector_type(4)))  unsigned v4u;
typedef __attribute__((ext_vector_type(8)))  int      v8i;
typedef __attribute__((ext_vector_type(4)))  int      v4i;

__device__ __forceinline__ v8f wmma16(v16h a, v16h b, v8f c) {
  // D = A(16x32 f16) x B(32x16 f16) + C(16x16 f32)
  return __builtin_amdgcn_wmma_f32_16x16x32_f16(false, a, false, b, (short)0, c,
                                                false, false);
}

__device__ __forceinline__ v8f zero8() {
  v8f z;
#pragma unroll
  for (int t = 0; t < 8; ++t) z[t] = 0.0f;
  return z;
}

// order-preserving float<->uint for atomicMax-based segment max
__device__ __forceinline__ unsigned fenc(float f) {
  unsigned u = __float_as_uint(f);
  return (u & 0x80000000u) ? ~u : (u | 0x80000000u);
}
__device__ __forceinline__ float fdec(unsigned u) {
  return __uint_as_float((u & 0x80000000u) ? (u & 0x7FFFFFFFu) : ~u);
}

// ---------------------------------------------------------------------------
// Pre-swizzle a [K x 128] f32 weight into per-lane WMMA B-fragment layout:
// dst[((kt*8+n)*32 + lane)*16 + j] = W[(kt*32 + (lane>>4)*8 + j + ((j>>3)<<3))*128 + n*16 + (lane&15)]
// so each fragment is one contiguous 32B load per lane.
// ---------------------------------------------------------------------------
__global__ void pack_w(const float* __restrict__ src, _Float16* __restrict__ dst,
                       int K) {
  int idx = blockIdx.x * blockDim.x + threadIdx.x;
  if (idx >= K * DIM) return;
  int j    = idx & 15;
  int lane = (idx >> 4) & 31;
  int n    = (idx >> 9) & 7;
  int kt   = idx >> 12;
  int col  = lane & 15;
  int kb   = (lane >> 4) * 8;
  int srcK = kt * 32 + kb + j + ((j >> 3) << 3);
  dst[idx] = (_Float16)src[srcK * DIM + n * 16 + col];
}

__global__ void embed_k(const float* __restrict__ emb, const int* __restrict__ z,
                        float* __restrict__ x) {
  int idx = blockIdx.x * blockDim.x + threadIdx.x;  // v4f granules
  if (idx >= N_ATOMS * DIM / 4) return;
  int n = idx >> 5, d4 = idx & 31;
  ((v4f*)x)[idx] = ((const v4f*)emb)[z[n] * 32 + d4];
}

__global__ void zero_k(float* __restrict__ p, int n) {
  int stride = gridDim.x * blockDim.x;
  for (int i = blockIdx.x * blockDim.x + threadIdx.x; i < n; i += stride)
    p[i] = 0.0f;
}

// ---------------------------------------------------------------------------
// Edge filter: Wfil = ((f@Wl + bl) + (swish(f@W1f + b1f)@W2f + b2f)) * C(r)
// One wave per 16-edge tile. f_ij built directly in A-fragment layout.
// Intermediate u transposed D-layout -> A-layout through LDS.
// ---------------------------------------------------------------------------
__global__ __launch_bounds__(256) void filter_k(
    const float* __restrict__ xyz, const int* __restrict__ src,
    const int* __restrict__ dst, const _Float16* __restrict__ WlH,
    const _Float16* __restrict__ W1fH, const _Float16* __restrict__ W2fH,
    const float* __restrict__ bl, const float* __restrict__ b1f,
    const float* __restrict__ b2f, _Float16* __restrict__ WfilH) {
  __shared__ _Float16 uLDS[8 * 16 * DIM];
  int lane = threadIdx.x & 31;
  int wave = threadIdx.x >> 5;
  int tile = blockIdx.x * 8 + wave;
  int e0 = tile * 16;
  int row = lane & 15;
  int hi = lane >> 4;
  int kbA = hi * 8;
  int col = row;

  int e = e0 + row;
  int si = src[e], di = dst[e];
  float dx = xyz[si * 3 + 0] - xyz[di * 3 + 0];
  float dy = xyz[si * 3 + 1] - xyz[di * 3 + 1];
  float dz = xyz[si * 3 + 2] - xyz[di * 3 + 2];
  float r = sqrtf(dx * dx + dy * dy + dz * dz + 1e-12f);
  float Cc = (r < 8.0f) ? 0.5f * (cosf(r * PI_F / 8.0f) + 1.0f) : 0.0f;

  // Gaussian basis A-fragment (16x32), row = lane&15
  const float width = 8.0f / 31.0f;
  const float c2 = 0.5f / (width * width);
  v16h af;
#pragma unroll
  for (int j = 0; j < 16; ++j) {
    int K = kbA + j + ((j >> 3) << 3);
    float dd = r - (float)K * width;
    af[j] = (_Float16)__expf(-c2 * dd * dd);
  }

  _Float16* myU = uLDS + wave * 16 * DIM;
  // u = swish(f @ W1f + b1f)  -> LDS (D-layout write)
#pragma unroll
  for (int n = 0; n < 8; ++n) {
    v16h b = *(const v16h*)(W1fH + (n << 9) + (lane << 4));
    v8f acc = wmma16(af, b, zero8());
    float bb = b1f[n * 16 + col];
#pragma unroll
    for (int rr = 0; rr < 8; ++rr) {
      float u = acc[rr] + bb;
      float us = u / (1.0f + __expf(-u));
      myU[(rr + hi * 8) * DIM + n * 16 + col] = (_Float16)us;
    }
  }
  __syncthreads();

  // out = f@Wl + u@W2f + biases, times cutoff
#pragma unroll
  for (int n = 0; n < 8; ++n) {
    v16h bL = *(const v16h*)(WlH + (n << 9) + (lane << 4));
    v8f acc = wmma16(af, bL, zero8());
#pragma unroll
    for (int kt = 0; kt < 4; ++kt) {
      const _Float16* up = myU + row * DIM + kt * 32;
      v8h u0 = *(const v8h*)(up + kbA);        // ds_load_b128
      v8h u1 = *(const v8h*)(up + kbA + 16);
      v16h a2 = __builtin_shufflevector(u0, u1, 0, 1, 2, 3, 4, 5, 6, 7, 8, 9,
                                        10, 11, 12, 13, 14, 15);
      v16h b2 = *(const v16h*)(W2fH + ((kt * 8 + n) << 9) + (lane << 4));
      acc = wmma16(a2, b2, acc);
    }
    int c = n * 16 + col;
    float bb = bl[c] + b2f[c];
#pragma unroll
    for (int rr = 0; rr < 8; ++rr) {
      int M = rr + hi * 8;
      float Cm = __shfl(Cc, M);
      WfilH[(size_t)(e0 + M) * DIM + c] = (_Float16)((acc[rr] + bb) * Cm);
    }
  }
}

// ---------------------------------------------------------------------------
// Node pass: h = x + temb; q = h@Wq; k = h@Wk; store h as f16 for edge gathers.
// ---------------------------------------------------------------------------
__global__ __launch_bounds__(256) void node_qk_k(
    const float* __restrict__ x, const float* __restrict__ temb,
    const _Float16* __restrict__ WqH, const _Float16* __restrict__ WkH,
    _Float16* __restrict__ hH, float* __restrict__ q, float* __restrict__ k) {
  int lane = threadIdx.x & 31;
  int wave = threadIdx.x >> 5;
  int tile = blockIdx.x * 8 + wave;
  if (tile >= NT_NODE) return;
  int n0 = tile * 16;
  int row = lane & 15, hi = lane >> 4, kbA = hi * 8, col = row;

  // h -> f16 (vectorized: 4 elems per lane-iter)
#pragma unroll 4
  for (int it = 0; it < 16; ++it) {
    int c4 = (it * 32 + lane) * 4;
    int rr = c4 >> 7, c = c4 & 127;
    v4f xv = *(const v4f*)(x + (size_t)(n0 + rr) * DIM + c);
    v4f tv = *(const v4f*)(temb + c);
    v4h hv;
#pragma unroll
    for (int t = 0; t < 4; ++t) hv[t] = (_Float16)(xv[t] + tv[t]);
    *(v4h*)(hH + (size_t)(n0 + rr) * DIM + c) = hv;
  }

  v8f qa[8], ka[8];
#pragma unroll
  for (int n = 0; n < 8; ++n) { qa[n] = zero8(); ka[n] = zero8(); }

  const float* xr = x + (size_t)(n0 + row) * DIM;
#pragma unroll
  for (int kt = 0; kt < 4; ++kt) {
    int kk = kt * 32;
    v4f x0 = *(const v4f*)(xr + kk + kbA);
    v4f x1 = *(const v4f*)(xr + kk + kbA + 4);
    v4f x2 = *(const v4f*)(xr + kk + kbA + 16);
    v4f x3 = *(const v4f*)(xr + kk + kbA + 20);
    v4f t0 = *(const v4f*)(temb + kk + kbA);
    v4f t1 = *(const v4f*)(temb + kk + kbA + 4);
    v4f t2 = *(const v4f*)(temb + kk + kbA + 16);
    v4f t3 = *(const v4f*)(temb + kk + kbA + 20);
    v16h a;
#pragma unroll
    for (int t = 0; t < 4; ++t) {
      a[t]      = (_Float16)(x0[t] + t0[t]);
      a[4 + t]  = (_Float16)(x1[t] + t1[t]);
      a[8 + t]  = (_Float16)(x2[t] + t2[t]);
      a[12 + t] = (_Float16)(x3[t] + t3[t]);
    }
#pragma unroll
    for (int n = 0; n < 8; ++n) {
      v16h bq = *(const v16h*)(WqH + ((kt * 8 + n) << 9) + (lane << 4));
      v16h bk = *(const v16h*)(WkH + ((kt * 8 + n) << 9) + (lane << 4));
      qa[n] = wmma16(a, bq, qa[n]);
      ka[n] = wmma16(a, bk, ka[n]);
    }
  }
#pragma unroll
  for (int n = 0; n < 8; ++n) {
    int c = n * 16 + col;
#pragma unroll
    for (int rr = 0; rr < 8; ++rr) {
      size_t o = (size_t)(n0 + rr + hi * 8) * DIM + c;
      q[o] = qa[n][rr];
      k[o] = ka[n][rr];
    }
  }
}

// ---------------------------------------------------------------------------
// Edge pass 1: v = (h[src] .* Wfil) @ Wv (WMMA); s = scale * <q[dst],k[src]>;
// segment max via order-preserving uint atomicMax.
// Wv B-fragments (32KB) staged into LDS once per workgroup by the TDM
// (tensor_load_to_lds + s_wait_tensorcnt), then read as ds_load_b128.
// ---------------------------------------------------------------------------
__global__ __launch_bounds__(256) void edge_v_k(
    const _Float16* __restrict__ hH, const _Float16* __restrict__ WfilH,
    const _Float16* __restrict__ WvH, const int* __restrict__ src,
    const int* __restrict__ dst, const float* __restrict__ q,
    const float* __restrict__ k, _Float16* __restrict__ vH,
    float* __restrict__ sArr, unsigned* __restrict__ smaxU) {
  __shared__ _Float16 wlds[32 * 512];  // 32 B-fragments x 512 halves = 32KB
  int lane = threadIdx.x & 31;
  int wave = threadIdx.x >> 5;
  int tile = blockIdx.x * 8 + wave;
  int e0 = tile * 16;
  int row = lane & 15, hi = lane >> 4, kbA = hi * 8, col = row;
  int e = e0 + row;
  int si = src[e];

  // ---- TDM: DMA the packed Wv fragment block (linear 16KB) into LDS ----
  if (wave == 0) {
    unsigned long long ga = (unsigned long long)(const void*)WvH;
    v4u g0;
    g0[0] = 1u;                                   // count=1, user mode
    g0[1] = 0u;                                   // lds_addr = 0 (wlds base)
    g0[2] = (unsigned)(ga & 0xFFFFFFFFu);         // global_addr[31:0]
    g0[3] = (unsigned)((ga >> 32) & 0x1FFFFFFu) | (2u << 30);  // addr hi | type=2
    v8i g1;
    g1[0] = (int)(2u << 16);      // workgroup_mask=0, data_size=2 (4 bytes)
    g1[1] = (int)(4096u << 16);   // tensor_dim0 = 4096 dwords (low 16 bits)
    g1[2] = (int)(1u << 16);      // tensor_dim0 hi=0 | tensor_dim1 = 1
    g1[3] = (int)(4096u << 16);   // tensor_dim1 hi=0 | tile_dim0 = 4096
    g1[4] = 1;                    // tile_dim1 = 1, tile_dim2 = 0
    g1[5] = 4096;                 // tensor_dim0_stride = 4096 dwords
    g1[6] = 0;
    g1[7] = 0;
    v4i gz4;
    gz4[0] = 0; gz4[1] = 0; gz4[2] = 0; gz4[3] = 0;  // groups 2/3 unused (2D)
    v8i gz8;
#pragma unroll
    for (int t = 0; t < 8; ++t) gz8[t] = 0;
    __builtin_amdgcn_tensor_load_to_lds(g0, g1, gz4, gz4, gz8, 0);
    __builtin_amdgcn_s_wait_tensorcnt(0);
  }
  __syncthreads();
  const v16h* wfrag = (const v16h*)wlds;

  const _Float16* hr = hH + (size_t)si * DIM;
  const _Float16* fr = WfilH + (size_t)e * DIM;
  if (e + 16 < N_EDGES)
    __builtin_prefetch(fr + 16 * DIM, 0, 1);  // next tile's Wfil row

  v8f va[8];
#pragma unroll
  for (int n = 0; n < 8; ++n) va[n] = zero8();

#pragma unroll
  for (int kt = 0; kt < 4; ++kt) {
    int kk = kt * 32;
    v8h h0 = *(const v8h*)(hr + kk + kbA);       // global_load_b128
    v8h h1 = *(const v8h*)(hr + kk + kbA + 16);
    v8h f0 = *(const v8h*)(fr + kk + kbA);
    v8h f1 = *(const v8h*)(fr + kk + kbA + 16);
    v8h p0 = h0 * f0;                            // v_pk_mul_f16
    v8h p1 = h1 * f1;
    v16h a = __builtin_shufflevector(p0, p1, 0, 1, 2, 3, 4, 5, 6, 7, 8, 9, 10,
                                     11, 12, 13, 14, 15);
#pragma unroll
    for (int n = 0; n < 8; ++n) {
      v16h b = wfrag[(kt * 8 + n) * 32 + lane];  // ds_load_b128 x2
      va[n] = wmma16(a, b, va[n]);
    }
  }
#pragma unroll
  for (int n = 0; n < 8; ++n) {
    int c = n * 16 + col;
#pragma unroll
    for (int rr = 0; rr < 8; ++rr)
      vH[(size_t)(e0 + rr + hi * 8) * DIM + c] = (_Float16)va[n][rr];
  }

  // per-edge per-head scores (128 values per tile, 4 per lane)
#pragma unroll
  for (int rep = 0; rep < 4; ++rep) {
    int idx = lane + rep * 32;
    int el = idx >> 3, head = idx & 7;
    int e2 = e0 + el;
    int s2 = src[e2], d2 = dst[e2];
    const v4f* qp = (const v4f*)(q + (size_t)d2 * DIM + head * 16);
    const v4f* kp = (const v4f*)(k + (size_t)s2 * DIM + head * 16);
    float acc = 0.0f;
#pragma unroll
    for (int t = 0; t < 4; ++t) {
      v4f qq = qp[t], kk2 = kp[t];
      acc += qq[0] * kk2[0] + qq[1] * kk2[1] + qq[2] * kk2[2] + qq[3] * kk2[3];
    }
    acc *= 0.25f;  // 1/sqrt(dh=16)
    sArr[(size_t)e2 * NH + head] = acc;
    atomicMax(&smaxU[(size_t)d2 * NH + head], fenc(acc));
  }
}

// ---------------------------------------------------------------------------
// Edge pass 2: es = exp(s - smax[dst]); denom[dst] += es; msg[dst] += es * v
// One wave per edge; lanes 0..7 own heads, all lanes scatter 4 dims each.
// ---------------------------------------------------------------------------
__global__ __launch_bounds__(256) void edge_soft_k(
    const float* __restrict__ sArr, const unsigned* __restrict__ smaxU,
    const int* __restrict__ dst, const _Float16* __restrict__ vH,
    float* __restrict__ denom, float* __restrict__ msg) {
  int lane = threadIdx.x & 31;
  int wave = threadIdx.x >> 5;
  int e = blockIdx.x * 8 + wave;
  int d2 = dst[e];
  float es = 0.0f;
  if (lane < NH) {
    float s = sArr[(size_t)e * NH + lane];
    float m = fdec(smaxU[(size_t)d2 * NH + lane]);
    es = __expf(s - m);
    atomicAdd(&denom[(size_t)d2 * NH + lane], es);
  }
  float esb = __shfl(es, lane >> 2);  // head = dim>>4 = lane>>2
  v4h vv = *(const v4h*)(vH + (size_t)e * DIM + lane * 4);  // b64 load
  float* mp = msg + (size_t)d2 * DIM + lane * 4;
#pragma unroll
  for (int t = 0; t < 4; ++t) atomicAdd(&mp[t], esb * (float)vv[t]);
}

// ---------------------------------------------------------------------------
// Node pass 2: x += ((msg / denom_per_head) @ Wo)
// ---------------------------------------------------------------------------
__global__ __launch_bounds__(256) void node_out_k(
    const float* __restrict__ msg, const float* __restrict__ denom,
    const _Float16* __restrict__ WoH, float* __restrict__ x) {
  int lane = threadIdx.x & 31;
  int wave = threadIdx.x >> 5;
  int tile = blockIdx.x * 8 + wave;
  if (tile >= NT_NODE) return;
  int n0 = tile * 16;
  int row = lane & 15, hi = lane >> 4, kbA = hi * 8, col = row;

  const float* mr = msg + (size_t)(n0 + row) * DIM;
  float rd[8];
#pragma unroll
  for (int hh = 0; hh < 8; ++hh)
    rd[hh] = 1.0f / (denom[(size_t)(n0 + row) * NH + hh] + 1e-12f);

  v8f acc[8];
#pragma unroll
  for (int n = 0; n < 8; ++n) acc[n] = zero8();

#pragma unroll
  for (int kt = 0; kt < 4; ++kt) {
    int kk = kt * 32;
    v4f m0 = *(const v4f*)(mr + kk + kbA);
    v4f m1 = *(const v4f*)(mr + kk + kbA + 4);
    v4f m2 = *(const v4f*)(mr + kk + kbA + 16);
    v4f m3 = *(const v4f*)(mr + kk + kbA + 20);
    v16h a;
#pragma unroll
    for (int t = 0; t < 4; ++t) {
      a[t]      = (_Float16)(m0[t] * rd[(kk + kbA + t) >> 4]);
      a[4 + t]  = (_Float16)(m1[t] * rd[(kk + kbA + 4 + t) >> 4]);
      a[8 + t]  = (_Float16)(m2[t] * rd[(kk + kbA + 16 + t) >> 4]);
      a[12 + t] = (_Float16)(m3[t] * rd[(kk + kbA + 20 + t) >> 4]);
    }
#pragma unroll
    for (int n = 0; n < 8; ++n) {
      v16h b = *(const v16h*)(WoH + ((kt * 8 + n) << 9) + (lane << 4));
      acc[n] = wmma16(a, b, acc[n]);
    }
  }
#pragma unroll
  for (int n = 0; n < 8; ++n) {
    int c = n * 16 + col;
#pragma unroll
    for (int rr = 0; rr < 8; ++rr) {
      size_t o = (size_t)(n0 + rr + hi * 8) * DIM + c;
      x[o] += acc[n][rr];
    }
  }
}

// ---------------------------------------------------------------------------
extern "C" void kernel_launch(void* const* d_in, const int* in_sizes, int n_in,
                              void* d_out, int out_size, void* d_ws,
                              size_t ws_size, hipStream_t stream) {
  (void)in_sizes; (void)n_in; (void)out_size; (void)ws_size;
  const float* xyz  = (const float*)d_in[0];
  const float* emb  = (const float*)d_in[1];
  const float* temb = (const float*)d_in[2];
  const float* Wl   = (const float*)d_in[3];
  const float* bl   = (const float*)d_in[4];
  const float* W1f  = (const float*)d_in[5];
  const float* b1f  = (const float*)d_in[6];
  const float* W2f  = (const float*)d_in[7];
  const float* b2f  = (const float*)d_in[8];
  const float* Wq   = (const float*)d_in[9];
  const float* Wk   = (const float*)d_in[10];
  const float* Wv   = (const float*)d_in[11];
  const float* Wo   = (const float*)d_in[12];
  const int* z   = (const int*)d_in[13];
  const int* src = (const int*)d_in[14];
  const int* dst = (const int*)d_in[15];
  float* x = (float*)d_out;  // node features live in d_out

  char* w = (char*)d_ws;
  size_t off = 0;
  auto alloc = [&](size_t bytes) -> void* {
    void* p = (void*)(w + off);
    off = (off + bytes + 255) & ~(size_t)255;
    return p;
  };
  _Float16* hH    = (_Float16*)alloc((size_t)N_ATOMS * DIM * 2);
  float*    q     = (float*)   alloc((size_t)N_ATOMS * DIM * 4);
  float*    kbuf  = (float*)   alloc((size_t)N_ATOMS * DIM * 4);
  float*    msg   = (float*)   alloc((size_t)N_ATOMS * DIM * 4);
  float*    denom = (float*)   alloc((size_t)N_ATOMS * NH * 4);
  unsigned* smaxU = (unsigned*)alloc((size_t)N_ATOMS * NH * 4);
  float*    sArr  = (float*)   alloc((size_t)N_EDGES * NH * 4);
  _Float16* WfilH = (_Float16*)alloc((size_t)N_EDGES * DIM * 2);
  _Float16* vH    = (_Float16*)alloc((size_t)N_EDGES * DIM * 2);
  _Float16* WlH   = (_Float16*)alloc(NG * DIM * 2);
  _Float16* W1fH  = (_Float16*)alloc(NG * DIM * 2);
  _Float16* W2fH  = (_Float16*)alloc(DIM * DIM * 2);
  _Float16* WqH   = (_Float16*)alloc(DIM * DIM * 2);
  _Float16* WkH   = (_Float16*)alloc(DIM * DIM * 2);
  _Float16* WvH   = (_Float16*)alloc(DIM * DIM * 2);
  _Float16* WoH   = (_Float16*)alloc(DIM * DIM * 2);

  pack_w<<<(NG * DIM + 255) / 256, 256, 0, stream>>>(Wl, WlH, NG);
  pack_w<<<(NG * DIM + 255) / 256, 256, 0, stream>>>(W1f, W1fH, NG);
  pack_w<<<(DIM * DIM + 255) / 256, 256, 0, stream>>>(W2f, W2fH, DIM);
  pack_w<<<(DIM * DIM + 255) / 256, 256, 0, stream>>>(Wq, WqH, DIM);
  pack_w<<<(DIM * DIM + 255) / 256, 256, 0, stream>>>(Wk, WkH, DIM);
  pack_w<<<(DIM * DIM + 255) / 256, 256, 0, stream>>>(Wv, WvH, DIM);
  pack_w<<<(DIM * DIM + 255) / 256, 256, 0, stream>>>(Wo, WoH, DIM);

  embed_k<<<(N_ATOMS * DIM / 4 + 255) / 256, 256, 0, stream>>>(emb, z, x);
  filter_k<<<NT_EDGE / 8, 256, 0, stream>>>(xyz, src, dst, WlH, W1fH, W2fH, bl,
                                            b1f, b2f, WfilH);
  for (int i = 0; i < NINTER; ++i) {
    zero_k<<<512, 256, 0, stream>>>(msg, N_ATOMS * DIM);
    zero_k<<<64, 256, 0, stream>>>(denom, N_ATOMS * NH);
    zero_k<<<64, 256, 0, stream>>>((float*)smaxU, N_ATOMS * NH);  // 0 == -inf code
    node_qk_k<<<(NT_NODE + 7) / 8, 256, 0, stream>>>(x, temb + i * DIM, WqH,
                                                     WkH, hH, q, kbuf);
    edge_v_k<<<NT_EDGE / 8, 256, 0, stream>>>(hH, WfilH, WvH, src, dst, q, kbuf,
                                              vH, sArr, smaxU);
    edge_soft_k<<<N_EDGES / 8, 256, 0, stream>>>(sArr, smaxU, dst, vH, denom,
                                                 msg);
    node_out_k<<<(NT_NODE + 7) / 8, 256, 0, stream>>>(msg, denom, WoH, x);
  }
}
